// MultiHeadCrossAttention_11476152615205
// MI455X (gfx1250) — matrix operand order repaired and checked by reading, not compiled, not run on hardware
//
#include <hip/hip_runtime.h>
#include <hip/hip_bf16.h>

// ---------------------------------------------------------------------------
// MultiHeadCrossAttention for MI455X (gfx1250, wave32, WMMA bf16)
// Pipeline: cast->bf16, WMMA GEMM (QKV proj, async-LDS double buffered),
// per-head LN + layout, flash attention (WMMA + async-LDS K/V pipeline),
// final WMMA GEMM + bias (fp32 out).
// ---------------------------------------------------------------------------

typedef __attribute__((ext_vector_type(16))) __bf16 v16bf;
typedef __attribute__((ext_vector_type(8)))  __bf16 v8bf;
typedef __attribute__((ext_vector_type(4)))  __bf16 v4bf;
typedef __attribute__((ext_vector_type(8)))  float  v8f;

#define DEV __device__ __forceinline__

// Build a v16bf A/B fragment from two 16-byte LDS chunks.
DEV v16bf frag2(const __bf16* p0, const __bf16* p1) {
    v8bf a = *(const v8bf*)p0;
    v8bf b = *(const v8bf*)p1;
    v16bf r;
#pragma unroll
    for (int i = 0; i < 8; ++i) { r[i] = a[i]; r[i + 8] = b[i]; }
    return r;
}

DEV v8f wmma_bf16(v16bf a, v16bf b, v8f c) {
    return __builtin_amdgcn_wmma_f32_16x16x32_bf16(
        /*neg_a=*/false, a, /*neg_b=*/false, b,
        /*c_mod=*/(short)0, c, /*reuse_a=*/false, /*reuse_b=*/false);
}

// Async global -> LDS copy of 16 bytes (per lane). Tracked by ASYNCcnt.
// Generic pointers to __shared__ carry the LDS byte offset in addr[31:0].
DEV void async_b128(void* lds, const void* gaddr) {
    unsigned l = (unsigned)(size_t)lds;
    asm volatile("global_load_async_to_lds_b128 %0, %1, off"
                 :: "v"(l), "v"(gaddr) : "memory");
}

DEV void wait_async0() {
#if __has_builtin(__builtin_amdgcn_s_wait_asynccnt)
    __builtin_amdgcn_s_wait_asynccnt(0);
#else
    asm volatile("s_wait_asynccnt 0" ::: "memory");
#endif
}

// ---------------------------------------------------------------------------
// fp32 -> bf16 cast (vectorized by 4)
// ---------------------------------------------------------------------------
__global__ __launch_bounds__(256) void cast_f32_bf16(const float* __restrict__ in,
                                                     __bf16* __restrict__ out, int n4) {
    int i = blockIdx.x * 256 + threadIdx.x;
    if (i < n4) {
        float4 f = ((const float4*)in)[i];
        v4bf o;
        o[0] = (__bf16)f.x; o[1] = (__bf16)f.y; o[2] = (__bf16)f.z; o[3] = (__bf16)f.w;
        ((v4bf*)out)[i] = o;
    }
}

// ---------------------------------------------------------------------------
// WMMA GEMM:  C[M,N] = A[M,K] @ W[N,K]^T   (bf16 in, fp32 accum)
//  - FINAL=false: write bf16 C;   FINAL=true: write fp32 C + bias[n]
// 64x64 output tile, BK=64 (2 k-steps/barrier), 128 threads (4 waves),
// double-buffered async global->LDS staging.
// ---------------------------------------------------------------------------
#define LDSW 72   // 64 + 8 pad (144B row stride -> conflict-free 16-lane access)

template <bool FINAL>
__global__ __launch_bounds__(128) void wmma_gemm_nt(
    const __bf16* __restrict__ A, int lda,
    const __bf16* __restrict__ W, int ldw,
    __bf16* __restrict__ Cbf, float* __restrict__ Cf,
    const float* __restrict__ bias, int N, int K) {

    __shared__ __align__(16) __bf16 sA[2][64][LDSW];
    __shared__ __align__(16) __bf16 sB[2][64][LDSW];

    const int tid  = threadIdx.x;
    const int wave = tid >> 5;
    const int lane = tid & 31;
    const int half = lane >> 4;
    const int l16  = lane & 15;
    const int block_m = blockIdx.y * 64;
    const int block_n = blockIdx.x * 64;
    const int wm = (wave >> 1) * 32;
    const int wn = (wave & 1) * 32;

    const int ldr = tid >> 1;            // row 0..63
    const int ldc = (tid & 1) * 32;      // col 0 or 32

    v8f acc[2][2] = {};

    auto stage = [&](int buf, int k0) {
        const __bf16* gA = A + (size_t)(block_m + ldr) * lda + k0 + ldc;
        const __bf16* gW = W + (size_t)(block_n + ldr) * ldw + k0 + ldc;
#pragma unroll
        for (int j = 0; j < 4; ++j) {
            async_b128(&sA[buf][ldr][ldc + 8 * j], gA + 8 * j);
            async_b128(&sB[buf][ldr][ldc + 8 * j], gW + 8 * j);
        }
    };

    int cur = 0;
    stage(0, 0);
    wait_async0();
    __syncthreads();

    for (int k0 = 0; k0 < K; k0 += 64) {
        if (k0 + 64 < K) stage(cur ^ 1, k0 + 64);  // overlap with compute

#pragma unroll
        for (int ks = 0; ks < 2; ++ks) {
            v16bf af[2], bf[2];
#pragma unroll
            for (int mt = 0; mt < 2; ++mt) {
                // A 16x32: half0 -> K0..7|K16..23 ; half1 -> K8..15|K24..31
                const __bf16* row = &sA[cur][wm + mt * 16 + l16][ks * 32];
                af[mt] = frag2(row + half * 8, row + half * 8 + 16);
            }
#pragma unroll
            for (int nt = 0; nt < 2; ++nt) {
                // B 32x16: lane=N, half0 -> K0..7|K16..23 ; half1 -> K8..15|K24..31
                const __bf16* row = &sB[cur][wn + nt * 16 + l16][ks * 32];
                bf[nt] = frag2(row + half * 8, row + half * 8 + 16);
            }
#pragma unroll
            for (int mt = 0; mt < 2; ++mt)
#pragma unroll
                for (int nt = 0; nt < 2; ++nt)
                    acc[mt][nt] = wmma_bf16(af[mt], bf[nt], acc[mt][nt]);
        }

        wait_async0();
        __syncthreads();
        cur ^= 1;
    }

    // C layout: element r of accumulator -> (M = r + 8*half, N = l16)
#pragma unroll
    for (int mt = 0; mt < 2; ++mt)
#pragma unroll
        for (int nt = 0; nt < 2; ++nt) {
            int n = block_n + wn + nt * 16 + l16;
#pragma unroll
            for (int r = 0; r < 8; ++r) {
                int m = block_m + wm + mt * 16 + r + half * 8;
                if (FINAL)
                    Cf[(size_t)m * N + n] = acc[mt][nt][r] + bias[n];
                else
                    Cbf[(size_t)m * N + n] = (__bf16)acc[mt][nt][r];
            }
        }
}

// ---------------------------------------------------------------------------
// Per-head LayerNorm over D=64 + relayout [B*S, H*64] -> [(b*H+h)*S + s, 64]
// One wave per (row, head). 8 waves / 256-thread block.
// ---------------------------------------------------------------------------
__global__ __launch_bounds__(256) void ln_head_kernel(
    const __bf16* __restrict__ in, __bf16* __restrict__ out,
    const float* __restrict__ w, const float* __restrict__ b, int S) {
    const int wave = threadIdx.x >> 5, lane = threadIdx.x & 31;
    const int gid = blockIdx.x * 8 + wave;
    const int m = gid >> 4;          // row in [0, B*S)
    const int h = gid & 15;
    const int bi = m / S, s = m - bi * S;

    const __bf16* src = in + (size_t)m * 1024 + h * 64;
    float x0 = (float)src[lane], x1 = (float)src[lane + 32];

    float sum = x0 + x1;
#pragma unroll
    for (int off = 16; off > 0; off >>= 1) sum += __shfl_xor(sum, off, 32);
    float mu = sum * (1.0f / 64.0f);
    float d0 = x0 - mu, d1 = x1 - mu;
    float vs = d0 * d0 + d1 * d1;
#pragma unroll
    for (int off = 16; off > 0; off >>= 1) vs += __shfl_xor(vs, off, 32);
    float rstd = rsqrtf(vs * (1.0f / 64.0f) + 1e-5f);

    __bf16* dst = out + ((size_t)(bi * 16 + h) * S + s) * 64;
    dst[lane]      = (__bf16)(d0 * rstd * w[lane] + b[lane]);
    dst[lane + 32] = (__bf16)(d1 * rstd * w[lane + 32] + b[lane + 32]);
}

// ---------------------------------------------------------------------------
// V relayout: [B, S, H*64] -> Vt[(b*H+h)*64 + d, S]  (coalesced writes)
// ---------------------------------------------------------------------------
__global__ __launch_bounds__(256) void vtrans_kernel(const __bf16* __restrict__ in,
                                                     __bf16* __restrict__ out) {
    size_t o = (size_t)blockIdx.x * 256 + threadIdx.x;  // over 2*16*64*2048
    int s  = (int)(o & 2047);
    int d  = (int)((o >> 11) & 63);
    int h  = (int)((o >> 17) & 15);
    int bb = (int)(o >> 21);
    out[o] = in[(((size_t)(bb * 2048 + s)) << 10) + h * 64 + d];
}

// ---------------------------------------------------------------------------
// Flash attention: one workgroup = (batch, head, 64-query block).
// Q[bh,SQ,64], K[bh,SK,64], Vt[bh,64,SK] bf16 -> O [B,SQ,H*64] bf16.
// 4 waves; wave w owns query rows w*16..w*16+15. fp32 online softmax.
// K/V chunks double-buffered via async global->LDS.
// ---------------------------------------------------------------------------
__global__ __launch_bounds__(128) void attn_kernel(
    const __bf16* __restrict__ Q, const __bf16* __restrict__ Kt,
    const __bf16* __restrict__ Vt, __bf16* __restrict__ O) {
    constexpr int SQc = 2048, SKc = 2048, Hc = 16;

    __shared__ __align__(16) __bf16 sQ[64][LDSW];
    __shared__ __align__(16) __bf16 sK[2][64][LDSW];
    __shared__ __align__(16) __bf16 sV[2][64][LDSW];      // [d][key]
    __shared__ __align__(16) __bf16 sP[4][16][LDSW];      // per-wave P staging

    const int tid  = threadIdx.x;
    const int wave = tid >> 5, lane = tid & 31;
    const int half = lane >> 4, l16 = lane & 15;
    const int q0 = blockIdx.x * 64;
    const int bh = blockIdx.z * Hc + blockIdx.y;

    const int ldr = tid >> 1;
    const int ldc = (tid & 1) * 32;

    auto stage_kv = [&](int buf, int s0) {
        const __bf16* gk = Kt + ((size_t)bh * SKc + s0 + ldr) * 64 + ldc;
        const __bf16* gv = Vt + ((size_t)bh * 64 + ldr) * SKc + s0 + ldc;
#pragma unroll
        for (int j = 0; j < 4; ++j) {
            async_b128(&sK[buf][ldr][ldc + 8 * j], gk + 8 * j);
            async_b128(&sV[buf][ldr][ldc + 8 * j], gv + 8 * j);
        }
    };

    // ---- stage first K/V chunk, load Q block synchronously in parallel ----
    int cur = 0;
    stage_kv(0, 0);
    {
        const __bf16* g = Q + ((size_t)bh * SQc + q0 + ldr) * 64 + ldc;
        *(v8bf*)&sQ[ldr][ldc]      = *(const v8bf*)(g);
        *(v8bf*)&sQ[ldr][ldc + 8]  = *(const v8bf*)(g + 8);
        *(v8bf*)&sQ[ldr][ldc + 16] = *(const v8bf*)(g + 16);
        *(v8bf*)&sQ[ldr][ldc + 24] = *(const v8bf*)(g + 24);
    }
    wait_async0();
    __syncthreads();

    v16bf qf[2];
    {
        const __bf16* row = &sQ[wave * 16 + l16][0];
        qf[0] = frag2(row + half * 8,      row + half * 8 + 16);
        qf[1] = frag2(row + 32 + half * 8, row + 32 + half * 8 + 16);
    }

    float m_i[8], l_i[8];
    v8f o_acc[4] = {};
#pragma unroll
    for (int r = 0; r < 8; ++r) { m_i[r] = -1e30f; l_i[r] = 0.0f; }
    const float scale = 0.125f;  // 1/sqrt(64)

    for (int s0 = 0; s0 < SKc; s0 += 64) {
        if (s0 + 64 < SKc) stage_kv(cur ^ 1, s0 + 64);  // overlap with compute

        // ---- S = Q @ K^T * scale : 4 key-tiles of 16 ----
        float st[4][8];
#pragma unroll
        for (int nt = 0; nt < 4; ++nt) {
            v8f sa = {};
            const __bf16* rowK = &sK[cur][nt * 16 + l16][0];
#pragma unroll
            for (int ks = 0; ks < 2; ++ks) {
                v16bf bf = frag2(rowK + ks * 32 + half * 8,
                                 rowK + ks * 32 + half * 8 + 16);
                sa = wmma_bf16(qf[ks], bf, sa);
            }
#pragma unroll
            for (int r = 0; r < 8; ++r) st[nt][r] = sa[r] * scale;
        }

        // ---- fp32 online softmax (row = r + 8*half, spread over 16 lanes) ----
#pragma unroll
        for (int r = 0; r < 8; ++r) {
            float mx = fmaxf(fmaxf(st[0][r], st[1][r]), fmaxf(st[2][r], st[3][r]));
            mx = fmaxf(mx, __shfl_xor(mx, 1, 32));
            mx = fmaxf(mx, __shfl_xor(mx, 2, 32));
            mx = fmaxf(mx, __shfl_xor(mx, 4, 32));
            mx = fmaxf(mx, __shfl_xor(mx, 8, 32));
            float mnew  = fmaxf(m_i[r], mx);
            float alpha = __expf(m_i[r] - mnew);
            float rsum = 0.0f;
#pragma unroll
            for (int nt = 0; nt < 4; ++nt) {
                float p = __expf(st[nt][r] - mnew);
                st[nt][r] = p;
                rsum += p;
            }
            rsum += __shfl_xor(rsum, 1, 32);
            rsum += __shfl_xor(rsum, 2, 32);
            rsum += __shfl_xor(rsum, 4, 32);
            rsum += __shfl_xor(rsum, 8, 32);
            l_i[r] = l_i[r] * alpha + rsum;
            m_i[r] = mnew;
#pragma unroll
            for (int dt = 0; dt < 4; ++dt) o_acc[dt][r] *= alpha;
        }

        // ---- stage P (C-layout -> LDS row-major) then O += P @ V ----
#pragma unroll
        for (int nt = 0; nt < 4; ++nt)
#pragma unroll
            for (int r = 0; r < 8; ++r)
                sP[wave][r + half * 8][nt * 16 + l16] = (__bf16)st[nt][r];
        __syncthreads();

        const __bf16* rowP = &sP[wave][l16][0];
#pragma unroll
        for (int ks = 0; ks < 2; ++ks) {
            v16bf pf = frag2(rowP + ks * 32 + half * 8,
                             rowP + ks * 32 + half * 8 + 16);
#pragma unroll
            for (int dt = 0; dt < 4; ++dt) {
                const __bf16* rowV = &sV[cur][dt * 16 + l16][0];
                v16bf vf = frag2(rowV + ks * 32 + half * 8,
                                 rowV + ks * 32 + half * 8 + 16);
                o_acc[dt] = wmma_bf16(pf, vf, o_acc[dt]);
            }
        }

        wait_async0();   // next chunk landed
        __syncthreads(); // everyone done reading current buffers
        cur ^= 1;
    }

    // ---- epilogue: O / l_i -> [B, SQ, H*64] bf16 ----
#pragma unroll
    for (int r = 0; r < 8; ++r) {
        float inv = 1.0f / l_i[r];
        int q = q0 + wave * 16 + r + half * 8;
        size_t base = ((size_t)blockIdx.z * SQc + q) * 1024 + blockIdx.y * 64;
#pragma unroll
        for (int dt = 0; dt < 4; ++dt)
            O[base + dt * 16 + l16] = (__bf16)(o_acc[dt][r] * inv);
    }
}

// ---------------------------------------------------------------------------
// Host-side launch
// ---------------------------------------------------------------------------
extern "C" void kernel_launch(void* const* d_in, const int* in_sizes, int n_in,
                              void* d_out, int out_size, void* d_ws, size_t ws_size,
                              hipStream_t stream) {
    constexpr int B = 2, SQ = 2048, SK = 2048, E = 1024, INNER = 1024, H = 16;
    const float* emb  = (const float*)d_in[0];
    const float* ctx  = (const float*)d_in[1];
    const float* Wq   = (const float*)d_in[2];
    const float* Wk   = (const float*)d_in[3];
    const float* Wv   = (const float*)d_in[4];
    const float* Wu   = (const float*)d_in[5];
    const float* bu   = (const float*)d_in[6];
    const float* qn_w = (const float*)d_in[7];
    const float* qn_b = (const float*)d_in[8];
    const float* kn_w = (const float*)d_in[9];
    const float* kn_b = (const float*)d_in[10];

    char* ws = (char*)d_ws;
    auto take = [&](size_t bytes) { char* p = ws; ws += bytes; return p; };

    const size_t actB = (size_t)B * SQ * E * 2;      // 8 MiB (bf16 activation)
    const size_t wB   = (size_t)INNER * E * 2;       // 2 MiB (bf16 weight)

    __bf16* embh  = (__bf16*)take(actB);
    __bf16* ctxh  = (__bf16*)take(actB);
    __bf16* Wqh   = (__bf16*)take(wB);
    __bf16* Wkh   = (__bf16*)take(wB);
    __bf16* Wvh   = (__bf16*)take(wB);
    __bf16* Wuh   = (__bf16*)take(wB);
    __bf16* q_raw = (__bf16*)take(actB);
    __bf16* k_raw = (__bf16*)take(actB);
    __bf16* v_raw = (__bf16*)take(actB);
    __bf16* qh    = (__bf16*)take(actB);
    __bf16* kh    = (__bf16*)take(actB);
    __bf16* vt    = (__bf16*)take(actB);
    __bf16* attnh = (__bf16*)take(actB);

    const int nAct = B * SQ * E, nW = INNER * E;
    cast_f32_bf16<<<(nAct / 4 + 255) / 256, 256, 0, stream>>>(emb, embh, nAct / 4);
    cast_f32_bf16<<<(nAct / 4 + 255) / 256, 256, 0, stream>>>(ctx, ctxh, nAct / 4);
    cast_f32_bf16<<<(nW / 4 + 255) / 256, 256, 0, stream>>>(Wq, Wqh, nW / 4);
    cast_f32_bf16<<<(nW / 4 + 255) / 256, 256, 0, stream>>>(Wk, Wkh, nW / 4);
    cast_f32_bf16<<<(nW / 4 + 255) / 256, 256, 0, stream>>>(Wv, Wvh, nW / 4);
    cast_f32_bf16<<<(nW / 4 + 255) / 256, 256, 0, stream>>>(Wu, Wuh, nW / 4);

    const int M = B * SQ;  // 4096
    dim3 ggrid(INNER / 64, M / 64);
    wmma_gemm_nt<false><<<ggrid, 128, 0, stream>>>(embh, E, Wqh, E, q_raw, nullptr, nullptr, INNER, E);
    wmma_gemm_nt<false><<<ggrid, 128, 0, stream>>>(ctxh, E, Wkh, E, k_raw, nullptr, nullptr, INNER, E);
    wmma_gemm_nt<false><<<ggrid, 128, 0, stream>>>(ctxh, E, Wvh, E, v_raw, nullptr, nullptr, INNER, E);

    ln_head_kernel<<<(M * H) / 8, 256, 0, stream>>>(q_raw, qh, qn_w, qn_b, SQ);
    ln_head_kernel<<<(M * H) / 8, 256, 0, stream>>>(k_raw, kh, kn_w, kn_b, SK);
    vtrans_kernel<<<(B * H * 64 * SK) / 256, 256, 0, stream>>>(v_raw, vt);

    dim3 agrid(SQ / 64, H, B);
    attn_kernel<<<agrid, 128, 0, stream>>>(qh, kh, vt, attnh);

    wmma_gemm_nt<true><<<dim3(E / 64, M / 64), 128, 0, stream>>>(
        attnh, INNER, Wuh, INNER, nullptr, (float*)d_out, bu, E, INNER);
}
